// QuantizeEMAReset2D_79963701117562
// MI455X (gfx1250) — compile-verified, matrix operand order checked by reading, not run on hardware
//
#include <hip/hip_runtime.h>
#include <hip/hip_bf16.h>

// ---------------- types ----------------
typedef __attribute__((ext_vector_type(16))) __bf16       v16bf;
typedef __attribute__((ext_vector_type(8)))  float        v8f;
typedef __attribute__((ext_vector_type(8)))  unsigned int v8u;
typedef __attribute__((ext_vector_type(4)))  unsigned int v4u;
typedef __attribute__((ext_vector_type(4)))  int          v4i;
typedef __attribute__((ext_vector_type(8)))  int          v8i;

// ---------------- problem constants ----------------
constexpr int BD = 32, CD = 512, JD = 22, TD = 64;
constexpr int NROWS = BD * JD * TD;      // 45056 flattened vectors
constexpr int NB    = 1024;              // codes
constexpr int NTILE = NROWS / 16;        // 2816 row tiles
constexpr int KB    = CD / 32;           // 16 k-blocks of 32
constexpr float MUC  = 0.99f;
constexpr float EPSC = 1e-7f;

// B slab geometry: per kb, all 1024 codes x 32 k-values of bf16 = 64 KB
constexpr unsigned BSLAB   = 65536;      // bytes per precision per kb
constexpr unsigned STAGE   = 2 * BSLAB;  // hi + lo = 128 KB
constexpr unsigned LDS_DYN = 2 * STAGE;  // double buffered = 256 KB (<320 KB WGP LDS)

// output layout (floats, concatenated in reference return order)
constexpr int O_XQ = 0;                           // 23,068,672
constexpr int O_IDX = O_XQ + NROWS * CD;          // 45,056
constexpr int O_CL  = O_IDX + NROWS;              // 1
constexpr int O_PP  = O_CL + 1;                   // 1
constexpr int O_CB  = O_PP + 1;                   // 524,288
constexpr int O_CS  = O_CB + NB * CD;             // 524,288
constexpr int O_CC  = O_CS + NB * CD;             // 1,024

// ---------------- helpers ----------------
static __device__ __forceinline__ unsigned short f32_to_bf16_rne(float f) {
  unsigned u = __float_as_uint(f);
  unsigned r = u + 0x7FFFu + ((u >> 16) & 1u);
  return (unsigned short)(r >> 16);
}
static __device__ __forceinline__ float bf16_to_f32(unsigned short h) {
  return __uint_as_float(((unsigned)h) << 16);
}
// WMMA 16-bit fragment layout (A 16x32 and B 32x16 share the lane/element map):
// lanes 0-15 hold k%32 in [0..7]∪[16..23], lanes 16-31 hold [8..15]∪[24..31]
static __device__ __forceinline__ void frag_lane_j(int row15, int kk, int& lane, int& j) {
  if (kk < 8)        { lane = row15;      j = kk;      }
  else if (kk < 16)  { lane = 16 + row15; j = kk - 8;  }
  else if (kk < 24)  { lane = row15;      j = kk - 8;  }
  else               { lane = 16 + row15; j = kk - 16; }
}
static __device__ __forceinline__ v16bf as_v16bf(v8u v) {
  return __builtin_bit_cast(v16bf, v);
}
static __device__ __forceinline__ v8f wmma_bf16(v16bf a, v16bf b, v8f c) {
  return __builtin_amdgcn_wmma_f32_16x16x32_bf16(false, a, false, b, (short)0, c,
                                                 false, false);
}

#if defined(__has_builtin)
#if __has_builtin(__builtin_amdgcn_tensor_load_to_lds) && \
    __has_builtin(__builtin_amdgcn_s_wait_tensorcnt)
#define VQ_USE_TDM 1
#endif
#endif
#ifndef VQ_USE_TDM
#define VQ_USE_TDM 0
#endif

#if VQ_USE_TDM
// One 1-D TDM copy of 64 KB: 8192 elements of 8 bytes (tile_dim0 fits 16 bits).
// D# bit layout per CDNA5 ISA ch.8 (group0: count/lds_addr/global_addr/type=2;
// group1: data_size=3(8B), tensor_dim0=tile_dim0=8192, dim0_stride=8192).
// This toolchain exposes the 6-arg builtin form:
//   (uint32x4 g0, int32x8 g1, int32x4, int32x4, int32x8, i32 cpol)
static __device__ __forceinline__ void tdm_load_64k(unsigned lds_byte_addr,
                                                    const void* gsrc) {
  unsigned long long ga = (unsigned long long)gsrc;
  v4u g0;
  g0.x = 1u;                                               // count=1, user mode
  g0.y = lds_byte_addr;                                    // lds_addr
  g0.z = (unsigned)(ga & 0xFFFFFFFFu);                     // global_addr[31:0]
  g0.w = (unsigned)((ga >> 32) & 0x01FFFFFFu) | (2u << 30);// addr[56:32], type=2
  v8i g1;
  g1[0] = (int)(3u << 16);       // wg_mask=0, data_size=3 (8 bytes)
  g1[1] = (int)(8192u << 16);    // tensor_dim0[15:0] at bits[63:48]
  g1[2] = (int)(1u << 16);       // tensor_dim0[31:16]=0, tensor_dim1=1
  g1[3] = (int)(8192u << 16);    // tensor_dim1 hi=0, tile_dim0=8192
  g1[4] = 0;                     // tile_dim1=0 (unused), tile_dim2=0
  g1[5] = 8192;                  // tensor_dim0_stride[31:0]
  g1[6] = 0;                     // stride hi, tensor_dim1_stride lo
  g1[7] = 0;
  v4i z4 = (v4i)0;
  v8i z8 = (v8i)0;
  __builtin_amdgcn_tensor_load_to_lds(g0, g1, z4, z4, z8, 0);
}
#endif

// ---------------- kernels ----------------
// (B,C,J,T) -> fragment-packed bf16 hi/lo of xf (N,C), layout [tile][kb][lane][j]
__global__ void vq_pack_x(const float* __restrict__ x,
                          unsigned short* __restrict__ Ah,
                          unsigned short* __restrict__ Al) {
  int gid = blockIdx.x * blockDim.x + threadIdx.x;   // over NROWS*CD, exact grid
  int n = gid >> 9, k = gid & 511;
  int b = n / (JD * TD), rem = n % (JD * TD), jj = rem >> 6, t = rem & 63;
  float v = x[((b * CD + k) * JD + jj) * TD + t];
  unsigned short hi = f32_to_bf16_rne(v);
  unsigned short lo = f32_to_bf16_rne(v - bf16_to_f32(hi));
  int lane, j;
  frag_lane_j(n & 15, k & 31, lane, j);
  int off = (((n >> 4) * KB + (k >> 5)) * 32 + lane) * 16 + j;
  Ah[off] = hi;
  Al[off] = lo;
}

// codebook -> fragment-packed bf16 hi/lo, kb-major layout [kb][ct][lane][j]
// (so each kb slab is one contiguous 64 KB block: TDM-friendly)
__global__ void vq_pack_cb(const float* __restrict__ cb,
                           unsigned short* __restrict__ Bh,
                           unsigned short* __restrict__ Bl) {
  int gid = blockIdx.x * blockDim.x + threadIdx.x;   // over NB*CD, exact grid
  int code = gid >> 9, k = gid & 511;
  float v = cb[gid];
  unsigned short hi = f32_to_bf16_rne(v);
  unsigned short lo = f32_to_bf16_rne(v - bf16_to_f32(hi));
  int lane, j;
  frag_lane_j(code & 15, k & 31, lane, j);
  int off = (((k >> 5) * (NB / 16) + (code >> 4)) * 32 + lane) * 16 + j;
  Bh[off] = hi;
  Bl[off] = lo;
}

__global__ void vq_cnorm(const float* __restrict__ cb, float* __restrict__ cnorm) {
  __shared__ float red[256];
  int code = blockIdx.x;
  float a = cb[code * CD + threadIdx.x];
  float b = cb[code * CD + 256 + threadIdx.x];
  red[threadIdx.x] = a * a + b * b;
  __syncthreads();
  for (int s = 128; s > 0; s >>= 1) {
    if (threadIdx.x < (unsigned)s) red[threadIdx.x] += red[threadIdx.x + s];
    __syncthreads();
  }
  if (threadIdx.x == 0) cnorm[code] = red[0];
}

__global__ void vq_zero(float* __restrict__ cs, float* __restrict__ cc,
                        float* __restrict__ commit) {
  int gid = blockIdx.x * blockDim.x + threadIdx.x;
  if (gid < NB * CD) cs[gid] = 0.0f;
  if (gid < NB) cc[gid] = 0.0f;
  if (gid == 0) commit[0] = 0.0f;
}

// 16-row tile x 1024 codes; 8 waves, wave w owns code tiles [8w, 8w+8).
// B slabs are staged in LDS by the Tensor Data Mover, double buffered.
__global__ __launch_bounds__(256) void
vq_gemm_argmin(const unsigned short* __restrict__ Ah,
               const unsigned short* __restrict__ Al,
               const unsigned short* __restrict__ Bh,
               const unsigned short* __restrict__ Bl,
               const float* __restrict__ cnorm,
               int* __restrict__ idx_i32, float* __restrict__ idx_f) {
  extern __shared__ char smem[];   // 2 stages x (64KB B_hi + 64KB B_lo)
  const int tile = blockIdx.x;
  const int wave = threadIdx.x >> 5, lane = threadIdx.x & 31;

  const v8u* Ah8 = (const v8u*)Ah;
  const v8u* Al8 = (const v8u*)Al;

  v8f acc[8];
#pragma unroll
  for (int i = 0; i < 8; ++i) acc[i] = (v8f)(0.0f);

#if VQ_USE_TDM
  const unsigned lds_base = (unsigned)(unsigned long long)smem;
  if (wave == 0) {   // prologue: stage kb=0 into buffer 0
    tdm_load_64k(lds_base + 0,     (const char*)Bh);
    tdm_load_64k(lds_base + BSLAB, (const char*)Bl);
  }
#endif

#pragma unroll 1
  for (int kb = 0; kb < KB; ++kb) {
    const int cur = kb & 1;
#if VQ_USE_TDM
    if (wave == 0) {
      if (kb + 1 < KB) {
        // issue next stage, then wait until the current stage's 2 ops landed
        unsigned dst = lds_base + (unsigned)((kb + 1) & 1) * STAGE;
        tdm_load_64k(dst,         (const char*)Bh + (size_t)(kb + 1) * BSLAB);
        tdm_load_64k(dst + BSLAB, (const char*)Bl + (size_t)(kb + 1) * BSLAB);
        __builtin_amdgcn_s_wait_tensorcnt((short)2);
      } else {
        __builtin_amdgcn_s_wait_tensorcnt((short)0);
      }
    }
    __syncthreads();   // current stage visible to all waves
#endif
    int aidx = tile * (KB * 32) + kb * 32 + lane;     // v8u units (32B chunks)
    if (kb + 1 < KB) {
      __builtin_prefetch(&Ah8[aidx + 32], 0, 1);
      __builtin_prefetch(&Al8[aidx + 32], 0, 1);
    }
    v16bf a_h = as_v16bf(Ah8[aidx]);
    v16bf a_l = as_v16bf(Al8[aidx]);
#if VQ_USE_TDM
    const v8u* sh = (const v8u*)(smem + cur * STAGE);
    const v8u* sl = (const v8u*)(smem + cur * STAGE + BSLAB);
#else
    const v8u* sh = (const v8u*)Bh + (size_t)kb * (NB / 16) * 32;
    const v8u* sl = (const v8u*)Bl + (size_t)kb * (NB / 16) * 32;
#endif
#pragma unroll
    for (int i = 0; i < 8; ++i) {
      int ct = wave * 8 + i;
      v16bf b_h = as_v16bf(sh[ct * 32 + lane]);
      v16bf b_l = as_v16bf(sl[ct * 32 + lane]);
      acc[i] = wmma_bf16(a_h, b_h, acc[i]);   // hi*hi
      acc[i] = wmma_bf16(a_h, b_l, acc[i]);   // hi*lo
      acc[i] = wmma_bf16(a_l, b_h, acc[i]);   // lo*hi
    }
#if VQ_USE_TDM
    __syncthreads();   // all reads done before TDM overwrites this buffer
#endif
  }

  // distances: d = ||c||^2 - 2*(x.c)  (row norm constant -> drop for argmin)
  float bestd[8];
  int besti[8];
#pragma unroll
  for (int v = 0; v < 8; ++v) { bestd[v] = 3.4e38f; besti[v] = 0; }
#pragma unroll
  for (int i = 0; i < 8; ++i) {
    int code = wave * 128 + i * 16 + (lane & 15);
    float cn = cnorm[code];
#pragma unroll
    for (int v = 0; v < 8; ++v) {
      float d = cn - 2.0f * acc[i][v];
      if (d < bestd[v] || (d == bestd[v] && code < besti[v])) {
        bestd[v] = d; besti[v] = code;
      }
    }
  }
  // reduce across the 16 columns (lanes within each half hold the same rows)
#pragma unroll
  for (int m = 1; m < 16; m <<= 1) {
#pragma unroll
    for (int v = 0; v < 8; ++v) {
      float od = __shfl_xor(bestd[v], m, 32);
      int oi = __shfl_xor(besti[v], m, 32);
      if (od < bestd[v] || (od == bestd[v] && oi < besti[v])) {
        bestd[v] = od; besti[v] = oi;
      }
    }
  }

  __shared__ float sd[8][16];
  __shared__ int si[8][16];
  if ((lane & 15) == 0) {
    int rbase = (lane >> 4) * 8;   // lane0 -> rows 0..7, lane16 -> rows 8..15
#pragma unroll
    for (int v = 0; v < 8; ++v) { sd[wave][rbase + v] = bestd[v]; si[wave][rbase + v] = besti[v]; }
  }
  __syncthreads();
  if (threadIdx.x < 16) {
    int r = threadIdx.x;
    float bd = sd[0][r];
    int bi = si[0][r];
#pragma unroll
    for (int w = 1; w < 8; ++w) {   // waves cover ascending code ranges: strict <
      float od = sd[w][r];
      int oi = si[w][r];
      if (od < bd) { bd = od; bi = oi; }
    }
    int n = tile * 16 + r;
    idx_i32[n] = bi;
    idx_f[n] = (float)bi;
  }
}

__global__ void vq_gather_scatter(const float* __restrict__ x,
                                  const float* __restrict__ cb,
                                  const int* __restrict__ idx,
                                  float* __restrict__ xq,
                                  float* __restrict__ cs,
                                  float* __restrict__ cc,
                                  float* __restrict__ commit) {
  __shared__ float red[256];
  int gid = blockIdx.x * blockDim.x + threadIdx.x;   // over NROWS*CD, exact grid
  int n = gid >> 9, c = gid & 511;
  int b = n / (JD * TD), rem = n % (JD * TD), jj = rem >> 6, t = rem & 63;
  int xoff = ((b * CD + c) * JD + jj) * TD + t;
  int best = idx[n];
  float q = cb[best * CD + c];
  float xv = x[xoff];
  xq[xoff] = q;                                     // x_q: same layout as x
  atomicAdd(&cs[best * CD + c], xv);                // segment_sum (L2 atomics)
  if (c == 0) atomicAdd(&cc[best], 1.0f);
  float d = xv - q;
  red[threadIdx.x] = d * d;
  __syncthreads();
  for (int s = 128; s > 0; s >>= 1) {
    if (threadIdx.x < (unsigned)s) red[threadIdx.x] += red[threadIdx.x + s];
    __syncthreads();
  }
  if (threadIdx.x == 0) atomicAdd(commit, red[0]);
}

__global__ void vq_ema(const float* __restrict__ code_sum,
                       const float* __restrict__ code_count,
                       const float* __restrict__ cb,
                       const float* __restrict__ cs,
                       const float* __restrict__ cc,
                       float* __restrict__ out) {
  int gid = blockIdx.x * blockDim.x + threadIdx.x;   // over NB*CD
  int code = gid >> 9, c = gid & 511;
  float ncs = MUC * code_sum[gid] + (1.0f - MUC) * cs[gid];
  float ncc = MUC * code_count[code] + (1.0f - MUC) * cc[code];
  out[O_CS + gid] = ncs;
  out[O_CB + gid] = (ncc >= 1.0f) ? (ncs / ncc) : cb[gid];
  if (c == 0) out[O_CC + code] = ncc;
}

__global__ void vq_finalize(const float* __restrict__ cc,
                            const float* __restrict__ commit,
                            float* __restrict__ out) {
  __shared__ float red[256];
  float s = 0.0f;
  for (int code = threadIdx.x; code < NB; code += 256) {
    float p = cc[code] * (1.0f / (float)NROWS);     // sum(cc) == NROWS exactly
    s += p * logf(p + EPSC);
  }
  red[threadIdx.x] = s;
  __syncthreads();
  for (int sft = 128; sft > 0; sft >>= 1) {
    if (threadIdx.x < (unsigned)sft) red[threadIdx.x] += red[threadIdx.x + sft];
    __syncthreads();
  }
  if (threadIdx.x == 0) {
    out[O_PP] = expf(-red[0]);
    out[O_CL] = commit[0] * (1.0f / (float)(NROWS * CD));
  }
}

// ---------------- launcher ----------------
extern "C" void kernel_launch(void* const* d_in, const int* in_sizes, int n_in,
                              void* d_out, int out_size, void* d_ws, size_t ws_size,
                              hipStream_t stream) {
  const float* x          = (const float*)d_in[0];
  const float* codebook   = (const float*)d_in[1];
  const float* code_sum   = (const float*)d_in[2];
  const float* code_count = (const float*)d_in[3];
  float* out = (float*)d_out;

  // workspace carve-up (all offsets 256B aligned)
  char* w = (char*)d_ws;
  size_t o = 0;
  unsigned short* Ah = (unsigned short*)(w + o); o += (size_t)NROWS * CD * 2;  // 46.1 MB
  unsigned short* Al = (unsigned short*)(w + o); o += (size_t)NROWS * CD * 2;  // 46.1 MB
  unsigned short* Bh = (unsigned short*)(w + o); o += (size_t)NB * CD * 2;     // 1 MB
  unsigned short* Bl = (unsigned short*)(w + o); o += (size_t)NB * CD * 2;     // 1 MB
  float* cnorm  = (float*)(w + o); o += NB * 4;
  int*   idxbuf = (int*)(w + o);   o += (size_t)NROWS * 4;
  float* cs     = (float*)(w + o); o += (size_t)NB * CD * 4;                   // 2 MB
  float* cc     = (float*)(w + o); o += NB * 4;
  float* commit = (float*)(w + o); o += 256;
  (void)ws_size; (void)in_sizes; (void)n_in; (void)out_size;

  constexpr int TPB = 256;
  vq_pack_x<<<(NROWS * CD) / TPB, TPB, 0, stream>>>(x, Ah, Al);
  vq_pack_cb<<<(NB * CD) / TPB, TPB, 0, stream>>>(codebook, Bh, Bl);
  vq_cnorm<<<NB, TPB, 0, stream>>>(codebook, cnorm);
  vq_zero<<<(NB * CD + TPB - 1) / TPB, TPB, 0, stream>>>(cs, cc, commit);
  vq_gemm_argmin<<<NTILE, TPB, VQ_USE_TDM ? LDS_DYN : 0, stream>>>(
      Ah, Al, Bh, Bl, cnorm, idxbuf, out + O_IDX);
  vq_gather_scatter<<<(NROWS * CD) / TPB, TPB, 0, stream>>>(
      x, codebook, idxbuf, out + O_XQ, cs, cc, commit);
  vq_ema<<<(NB * CD) / TPB, TPB, 0, stream>>>(code_sum, code_count, codebook, cs,
                                              cc, out);
  vq_finalize<<<1, TPB, 0, stream>>>(cc, commit, out);
}
